// Model_31954556682346
// MI455X (gfx1250) — matrix operand order repaired and checked by reading, not compiled, not run on hardware
//
#include <hip/hip_runtime.h>
#include <hip/hip_bf16.h>

// Problem dims (compile-time constants from the reference)
#define Xd 1024
#define Hd 1024
#define Zd 256
#define Ad 64
#define Td 64
#define Bd 256
#define TB (Td * Bd)   // 16384

typedef __attribute__((ext_vector_type(16))) __bf16 v16bf;
typedef __attribute__((ext_vector_type(8)))  float  v8f;

// ---------------------------------------------------------------------------
// CDNA5 async global->LDS copy (16B per lane), tracked by ASYNCcnt.
// vdst operand = per-lane LDS byte address; vaddr = 64-bit global address.
// LDS byte address = low 32 bits of the generic pointer (ISA 10.2 aperture map).
// ---------------------------------------------------------------------------
__device__ __forceinline__ void async_copy_b128(const __bf16* gsrc, __bf16* ldst)
{
    unsigned lds            = (unsigned)(unsigned long long)(uintptr_t)ldst;
    unsigned long long gaddr = (unsigned long long)(uintptr_t)gsrc;
    asm volatile("global_load_async_to_lds_b128 %0, %1, off"
                 :: "v"(lds), "v"(gaddr) : "memory");
}

__device__ __forceinline__ void wait_asynccnt0()
{
    asm volatile("s_wait_asynccnt 0x0" ::: "memory");
}

// ---------------------------------------------------------------------------
// Generic bf16 WMMA GEMM:  C[M,N] = act( concat(A0[M,K0], A1[M,K1]) @ W[N,K]^T + bias )
// Block tile 128x256, 256 threads = 8 wave32 waves (2 x 4); each wave owns a
// 64x64 tile = 4x4 grid of v_wmma_f32_16x16x32_bf16 (16 WMMA / K-chunk / wave).
// Double-buffered LDS filled by GLOBAL_LOAD_ASYNC_TO_LDS_B128; one barrier/iter.
// Requires: M % 128 == 0, N % 256 == 0, K0/K1 % 32 == 0 (true at all call sites).
// act: 0 = none, 1 = ELU, 2 = softplus.  Writes Cf (fp32) and/or Cb (bf16).
// ---------------------------------------------------------------------------
__global__ __launch_bounds__(256)
void wmma_gemm(const __bf16* __restrict__ A0, int lda0, int K0,
               const __bf16* __restrict__ A1, int lda1, int K1,
               const __bf16* __restrict__ W,  const float* __restrict__ bias,
               float* __restrict__ Cf, __bf16* __restrict__ Cb,
               int N, int act)
{
    __shared__ __align__(64) __bf16 As[2][128][32];   // 16 KB
    __shared__ __align__(64) __bf16 Bs[2][256][32];   // 32 KB

    const int tid  = threadIdx.x;
    const int wave = tid >> 5;
    const int lane = tid & 31;
    const int half = lane >> 4;     // 0: lanes 0-15, 1: lanes 16-31
    const int lid  = lane & 15;
    const int wm   = wave >> 2;     // 0..1  (64 rows each)
    const int wn   = wave & 3;      // 0..3  (64 cols each)

    const int blockM = blockIdx.y * 128;
    const int blockN = blockIdx.x * 256;
    const int Ktot   = K0 + K1;

    v8f acc[4][4];
#pragma unroll
    for (int i = 0; i < 4; ++i)
#pragma unroll
        for (int j = 0; j < 4; ++j)
#pragma unroll
            for (int e = 0; e < 8; ++e) acc[i][j][e] = 0.f;

    // Staging: A tile 128x32 (32B/thread = 2 async b128); B tile 256x32 (64B/thread = 4).
    const int r  = tid >> 1;          // 0..127 : A tile row
    const int kh = (tid & 1) * 16;    // 0 or 16

    auto stage = [&](int k0, int buf) {
        const __bf16* srcA = (k0 < K0)
            ? (A0 + (size_t)(blockM + r) * lda0 + (k0 + kh))
            : (A1 + (size_t)(blockM + r) * lda1 + (k0 - K0 + kh));
        async_copy_b128(srcA,     &As[buf][r][kh]);
        async_copy_b128(srcA + 8, &As[buf][r][kh + 8]);

        const __bf16* srcW = W + (size_t)(blockN + tid) * Ktot + k0;
        async_copy_b128(srcW,      &Bs[buf][tid][0]);
        async_copy_b128(srcW + 8,  &Bs[buf][tid][8]);
        async_copy_b128(srcW + 16, &Bs[buf][tid][16]);
        async_copy_b128(srcW + 24, &Bs[buf][tid][24]);
    };

    const int nchunks = Ktot >> 5;
    stage(0, 0);

    for (int i = 0; i < nchunks; ++i) {
        const int cb = i & 1;
        wait_asynccnt0();       // own stage(i) complete
        __syncthreads();        // everyone's stage(i) landed; everyone done reading cb^1
        if (i + 1 < nchunks) stage((i + 1) << 5, cb ^ 1);   // overlaps with compute below

        // B fragments: lane holds column (wn*64 + nt*16 + lid), K = half*16 .. half*16+15
        v16bf bfrag[4];
#pragma unroll
        for (int nt = 0; nt < 4; ++nt) {
            const int n = wn * 64 + nt * 16 + lid;
            bfrag[nt] = *(const v16bf*)&Bs[cb][n][half * 16];
        }

#pragma unroll
        for (int mt = 0; mt < 4; ++mt) {
            // A fragment (contiguous k order, not the ISA table order):
            // lanes 0-15 hold K 0..15; lanes 16-31 hold K 16..31 (matches B fragment)
            const int m = wm * 64 + mt * 16 + lid;
            const __bf16* a0p = &As[cb][m][half * 16];
            const __bf16* a1p = &As[cb][m][half * 16 + 8];
            v16bf afrag;
#pragma unroll
            for (int j = 0; j < 8; ++j) { afrag[j] = a0p[j]; afrag[j + 8] = a1p[j]; }
#pragma unroll
            for (int nt = 0; nt < 4; ++nt) {
                acc[mt][nt] = __builtin_amdgcn_wmma_f32_16x16x32_bf16(
                    false, afrag, false, bfrag[nt], (short)0, acc[mt][nt], false, false);
            }
        }
    }

    // ---- epilogue: bias + activation, fp32 and/or bf16 stores ----
#pragma unroll
    for (int mt = 0; mt < 4; ++mt) {
#pragma unroll
        for (int nt = 0; nt < 4; ++nt) {
            const int gn = blockN + wn * 64 + nt * 16 + lid;
            const float bv  = bias ? bias[gn] : 0.f;
            const int  gm0 = blockM + wm * 64 + mt * 16 + half * 8;
#pragma unroll
            for (int i = 0; i < 8; ++i) {
                float c = acc[mt][nt][i] + bv;
                if (act == 1)      c = (c > 0.f) ? c : (__expf(c) - 1.f);           // ELU
                else if (act == 2) c = (c > 20.f) ? c : log1pf(__expf(c));          // softplus
                const size_t off = (size_t)(gm0 + i) * N + gn;
                if (Cf) Cf[off] = c;
                if (Cb) Cb[off] = (__bf16)c;
            }
        }
    }
}

// ---------------------------------------------------------------------------
// Elementwise helpers
// ---------------------------------------------------------------------------
__global__ void f2bf_kernel(const float* __restrict__ src, __bf16* __restrict__ dst, int n)
{
    int i = blockIdx.x * blockDim.x + threadIdx.x;
    if (i < n) dst[i] = (__bf16)src[i];
}

__global__ void zero_state_kernel(float* __restrict__ h, __bf16* __restrict__ bel0, int n)
{
    int i = blockIdx.x * blockDim.x + threadIdx.x;
    if (i < n) { h[i] = 0.f; bel0[i] = (__bf16)0.f; }
}

// zq = zq_mean + zq_std * eps ; write fp32 output + bf16 copy for transition GEMM
__global__ void zq_kernel(const float* __restrict__ zqm, const float* __restrict__ zqs,
                          const float* __restrict__ eps, float* __restrict__ zq_out,
                          __bf16* __restrict__ zq_bf, int n)
{
    int i = blockIdx.x * blockDim.x + threadIdx.x;
    if (i >= n) return;
    float v = zqm[i] + zqs[i] * eps[i];
    zq_out[i] = v;
    zq_bf[i]  = (__bf16)v;
}

// GRU gates: gi,gh are [B, 3H]; h is fp32 recurrent state; writes bf16 beliefs[t+1]
__global__ void gru_kernel(const float* __restrict__ gi, const float* __restrict__ gh,
                           float* __restrict__ h, __bf16* __restrict__ bel_next, int n)
{
    int idx = blockIdx.x * blockDim.x + threadIdx.x;
    if (idx >= n) return;
    const int b = idx >> 10;          // H = 1024
    const int j = idx & 1023;
    const float* gib = gi + (size_t)b * 3 * Hd;
    const float* ghb = gh + (size_t)b * 3 * Hd;
    const float ir = gib[j], iz = gib[Hd + j], in = gib[2 * Hd + j];
    const float hr = ghb[j], hz = ghb[Hd + j], hn = ghb[2 * Hd + j];
    const float rg = 1.f / (1.f + __expf(-(ir + hr)));
    const float zg = 1.f / (1.f + __expf(-(iz + hz)));
    const float ng = tanhf(in + rg * hn);
    const float hv = h[idx];
    const float hnew = (1.f - zg) * ng + zg * hv;
    h[idx] = hnew;
    bel_next[idx] = (__bf16)hnew;
}

// N=1 heads (rew mean/std, gamma logit): one wave32 per row, shuffle reduction
__global__ __launch_bounds__(256)
void head_kernel(const __bf16* __restrict__ hid,
                 const float* __restrict__ w0, const float* __restrict__ b0,
                 const float* __restrict__ w1, const float* __restrict__ b1,
                 float* __restrict__ out0, float* __restrict__ out1, int rows)
{
    const int wave = threadIdx.x >> 5;
    const int lane = threadIdx.x & 31;
    const int row  = blockIdx.x * 8 + wave;
    if (row >= rows) return;
    const __bf16* hr = hid + (size_t)row * Hd;
    float a0 = 0.f, a1 = 0.f;
    for (int j = lane; j < Hd; j += 32) {
        const float hv = (float)hr[j];
        a0 += hv * w0[j];
        if (w1) a1 += hv * w1[j];
    }
#pragma unroll
    for (int off = 16; off > 0; off >>= 1) {
        a0 += __shfl_down(a0, off, 32);
        a1 += __shfl_down(a1, off, 32);
    }
    if (lane == 0) {
        out0[row] = a0 + b0[0];
        if (out1) {
            const float s = a1 + b1[0];
            out1[row] = (s > 20.f) ? s : log1pf(__expf(s));   // softplus
        }
    }
}

// ---------------------------------------------------------------------------
// Host orchestration
// ---------------------------------------------------------------------------
extern "C" void kernel_launch(void* const* d_in, const int* in_sizes, int n_in,
                              void* d_out, int out_size, void* d_ws, size_t ws_size,
                              hipStream_t stream)
{
    (void)in_sizes; (void)n_in; (void)out_size; (void)ws_size;

    // Inputs in setup_inputs() dict (insertion) order:
    const float* obs = (const float*)d_in[0];   // [T,B,X]
    const float* actf= (const float*)d_in[1];   // [T,B,A]
    const float* eps = (const float*)d_in[2];   // [T,B,Z]
    // zq_p: 3 W1, 4 b1, 5 Wm, 6 bm, 7 Ws, 8 bs
    // zp_p: 9..14 ; trans_p: 15 Wa,16 ba,17 Wih,18 bih,19 Whh,20 bhh
    // obsp_p: 21..26 ; rewp_p: 27..32 ; gammap_p: 33 W1,34 b1,35 W2,36 b2

    char* ws = (char*)d_ws;
    size_t cur = 0;
    auto alloc = [&](size_t bytes) -> void* {
        cur = (cur + 255) & ~(size_t)255;
        void* p = ws + cur;
        cur += bytes;
        return p;
    };
    auto allocBF = [&](size_t elems) -> __bf16* { return (__bf16*)alloc(elems * 2); };
    auto allocF  = [&](size_t elems) -> float*  { return (float*)alloc(elems * 4); };

    // bf16 weight copies
    __bf16* wq1 = allocBF((size_t)Hd * (Xd + Hd));
    __bf16* wqm = allocBF((size_t)Zd * Hd);
    __bf16* wqs = allocBF((size_t)Zd * Hd);
    __bf16* wp1 = allocBF((size_t)Hd * Hd);
    __bf16* wpm = allocBF((size_t)Zd * Hd);
    __bf16* wps = allocBF((size_t)Zd * Hd);
    __bf16* wa  = allocBF((size_t)Hd * (Zd + Ad));
    __bf16* wih = allocBF((size_t)3 * Hd * Hd);
    __bf16* whh = allocBF((size_t)3 * Hd * Hd);
    __bf16* wo1 = allocBF((size_t)Hd * (Zd + Hd));
    __bf16* wom = allocBF((size_t)Xd * Hd);
    __bf16* wos = allocBF((size_t)Xd * Hd);
    __bf16* wr1 = allocBF((size_t)Hd * (Zd + Hd));
    __bf16* wg1 = allocBF((size_t)Hd * (Zd + Hd));

    // bf16 activations / state
    __bf16* obs_bf = allocBF((size_t)TB * Xd);
    __bf16* act_bf = allocBF((size_t)TB * Ad);
    __bf16* bel_bf = allocBF((size_t)(Td + 1) * Bd * Hd);  // beliefs[0..T]
    __bf16* zq_bf  = allocBF((size_t)TB * Zd);
    __bf16* h1_bf  = allocBF((size_t)Bd * Hd);             // per-step hidden
    __bf16* u_bf   = allocBF((size_t)Bd * Hd);
    __bf16* hid_bf = allocBF((size_t)TB * Hd);             // post-scan hidden (reused 3x)
    float*  hf32   = allocF((size_t)Bd * Hd);              // fp32 recurrent h
    float*  gi_f   = allocF((size_t)Bd * 3 * Hd);
    float*  gh_f   = allocF((size_t)Bd * 3 * Hd);

    auto cvt = [&](const float* s, __bf16* d, size_t n) {
        f2bf_kernel<<<(unsigned)((n + 255) / 256), 256, 0, stream>>>(s, d, (int)n);
    };
    cvt((const float*)d_in[3],  wq1, (size_t)Hd * (Xd + Hd));
    cvt((const float*)d_in[5],  wqm, (size_t)Zd * Hd);
    cvt((const float*)d_in[7],  wqs, (size_t)Zd * Hd);
    cvt((const float*)d_in[9],  wp1, (size_t)Hd * Hd);
    cvt((const float*)d_in[11], wpm, (size_t)Zd * Hd);
    cvt((const float*)d_in[13], wps, (size_t)Zd * Hd);
    cvt((const float*)d_in[15], wa,  (size_t)Hd * (Zd + Ad));
    cvt((const float*)d_in[17], wih, (size_t)3 * Hd * Hd);
    cvt((const float*)d_in[19], whh, (size_t)3 * Hd * Hd);
    cvt((const float*)d_in[21], wo1, (size_t)Hd * (Zd + Hd));
    cvt((const float*)d_in[23], wom, (size_t)Xd * Hd);
    cvt((const float*)d_in[25], wos, (size_t)Xd * Hd);
    cvt((const float*)d_in[27], wr1, (size_t)Hd * (Zd + Hd));
    cvt((const float*)d_in[33], wg1, (size_t)Hd * (Zd + Hd));
    cvt(obs,  obs_bf, (size_t)TB * Xd);
    cvt(actf, act_bf, (size_t)TB * Ad);

    zero_state_kernel<<<(Bd * Hd) / 256, 256, 0, stream>>>(hf32, bel_bf, Bd * Hd);

    auto gemm = [&](const __bf16* A0, int lda0, int K0,
                    const __bf16* A1, int lda1, int K1,
                    const __bf16* W, const float* bias,
                    float* Cf, __bf16* Cb, int M, int N, int act) {
        dim3 g(N / 256, M / 128);
        wmma_gemm<<<g, 256, 0, stream>>>(A0, lda0, K0, A1, lda1, K1, W, bias, Cf, Cb, N, act);
    };

    float* out = (float*)d_out;
    const size_t BZ = (size_t)Bd * Zd;
    float* zq_out  = out;
    float* zqm_out = out + (size_t)TB * Zd;
    float* zqs_out = zqm_out + (size_t)TB * Zd;
    float* zpm_out = zqs_out + (size_t)TB * Zd;
    float* zps_out = zpm_out + (size_t)TB * Zd;
    float* obsm_out = zps_out + (size_t)TB * Zd;
    float* obss_out = obsm_out + (size_t)TB * Xd;
    float* rewm_out = obss_out + (size_t)TB * Xd;
    float* rews_out = rewm_out + TB;
    float* gam_out  = rews_out + TB;

    const float* b1q = (const float*)d_in[4];
    const float* bqm = (const float*)d_in[6];
    const float* bqs = (const float*)d_in[8];
    const float* bp1 = (const float*)d_in[10];
    const float* bpm = (const float*)d_in[12];
    const float* bps = (const float*)d_in[14];
    const float* ba  = (const float*)d_in[16];
    const float* bih = (const float*)d_in[18];
    const float* bhh = (const float*)d_in[20];
    const float* bo1 = (const float*)d_in[22];
    const float* bom = (const float*)d_in[24];
    const float* bos = (const float*)d_in[26];
    const float* br1 = (const float*)d_in[28];
    const float* bg1 = (const float*)d_in[34];

    // ---------------- sequential scan (stream-ordered; graph-capture safe) --------------
    for (int t = 0; t < Td; ++t) {
        const __bf16* ht = bel_bf + (size_t)t * Bd * Hd;

        // zq predictor: h1 = elu([obs_t, h] @ Wq1^T + b1)
        gemm(obs_bf + (size_t)t * Bd * Xd, Xd, Xd, ht, Hd, Hd,
             wq1, b1q, nullptr, h1_bf, Bd, Hd, 1);
        gemm(h1_bf, Hd, Hd, nullptr, 0, 0, wqm, bqm, zqm_out + t * BZ, nullptr, Bd, Zd, 0);
        gemm(h1_bf, Hd, Hd, nullptr, 0, 0, wqs, bqs, zqs_out + t * BZ, nullptr, Bd, Zd, 2);
        zq_kernel<<<(Bd * Zd) / 256, 256, 0, stream>>>(
            zqm_out + t * BZ, zqs_out + t * BZ, eps + t * BZ,
            zq_out + t * BZ, zq_bf + t * BZ, Bd * Zd);

        // zp predictor (prior): h1 = elu(h @ Wp1^T + b1)  (reuses h1_bf, stream ordered)
        gemm(ht, Hd, Hd, nullptr, 0, 0, wp1, bp1, nullptr, h1_bf, Bd, Hd, 1);
        gemm(h1_bf, Hd, Hd, nullptr, 0, 0, wpm, bpm, zpm_out + t * BZ, nullptr, Bd, Zd, 0);
        gemm(h1_bf, Hd, Hd, nullptr, 0, 0, wps, bps, zps_out + t * BZ, nullptr, Bd, Zd, 2);

        // transition: u = elu([zq, act] @ Wa^T + ba)
        gemm(zq_bf + t * BZ, Zd, Zd, act_bf + (size_t)t * Bd * Ad, Ad, Ad,
             wa, ba, nullptr, u_bf, Bd, Hd, 1);
        // GRU gates
        gemm(u_bf, Hd, Hd, nullptr, 0, 0, wih, bih, gi_f, nullptr, Bd, 3 * Hd, 0);
        gemm(ht,   Hd, Hd, nullptr, 0, 0, whh, bhh, gh_f, nullptr, Bd, 3 * Hd, 0);
        gru_kernel<<<(Bd * Hd) / 256, 256, 0, stream>>>(
            gi_f, gh_f, hf32, bel_bf + (size_t)(t + 1) * Bd * Hd, Bd * Hd);
    }

    // ---------------- post-scan parallel predictors (M = 16384) --------------
    const __bf16* bel_prev = bel_bf;                            // beliefs[0..T-1]
    const __bf16* bel_next = bel_bf + (size_t)Bd * Hd;          // beliefs[1..T]

    // obs predictor
    gemm(zq_bf, Zd, Zd, bel_prev, Hd, Hd, wo1, bo1, nullptr, hid_bf, TB, Hd, 1);
    gemm(hid_bf, Hd, Hd, nullptr, 0, 0, wom, bom, obsm_out, nullptr, TB, Xd, 0);
    gemm(hid_bf, Hd, Hd, nullptr, 0, 0, wos, bos, obss_out, nullptr, TB, Xd, 2);

    // reward predictor (N=1 heads via wave reduction)
    gemm(zq_bf, Zd, Zd, bel_next, Hd, Hd, wr1, br1, nullptr, hid_bf, TB, Hd, 1);
    head_kernel<<<TB / 8, 256, 0, stream>>>(
        hid_bf, (const float*)d_in[29], (const float*)d_in[30],
        (const float*)d_in[31], (const float*)d_in[32], rewm_out, rews_out, TB);

    // gamma predictor
    gemm(zq_bf, Zd, Zd, bel_next, Hd, Hd, wg1, bg1, nullptr, hid_bf, TB, Hd, 1);
    head_kernel<<<TB / 8, 256, 0, stream>>>(
        hid_bf, (const float*)d_in[35], (const float*)d_in[36],
        nullptr, nullptr, gam_out, nullptr, TB);
}
